// LLFQVAE_69483980915243
// MI455X (gfx1250) — matrix operand order repaired and checked by reading, not compiled, not run on hardware
//
#include <hip/hip_runtime.h>

#define B_SZ   32768
#define F_DIMC 512
#define HIDC   512
#define LATC   256
#define KCODES 4096

typedef __bf16       bf16x16 __attribute__((ext_vector_type(16)));
typedef float        f32x8   __attribute__((ext_vector_type(8)));
typedef unsigned int u32x8   __attribute__((ext_vector_type(8)));
typedef int          i32x4v  __attribute__((vector_size(16)));   // matches builtin param

union FragAB { bf16x16 b; u32x8 u; };

#if __has_builtin(__builtin_amdgcn_global_load_async_to_lds_b128)
#define HAVE_ASYNC_LDS 1
#else
#define HAVE_ASYNC_LDS 0
#endif

#define GLOBAL_AS __attribute__((address_space(1)))
#define LDS_AS    __attribute__((address_space(3)))

__device__ __forceinline__ unsigned short f2bf(float f) {
  unsigned int u = __float_as_uint(f);
  u += 0x7FFFu + ((u >> 16) & 1u);          // round-to-nearest-even
  return (unsigned short)(u >> 16);
}

__device__ __forceinline__ float gelu_exact(float v) {
  return 0.5f * v * (1.0f + erff(v * 0.70710678118654752440f));
}

// Stage 32 bytes (16 bf16) global -> LDS.  Async path uses the CDNA5
// GLOBAL_LOAD_ASYNC_TO_LDS_B128 (ASYNCcnt) and bypasses VGPRs.
__device__ __forceinline__ void stage32(const unsigned short* __restrict__ g,
                                        unsigned short* l)
{
#if HAVE_ASYNC_LDS
  __builtin_amdgcn_global_load_async_to_lds_b128(
      (GLOBAL_AS i32x4v*)(g),     (LDS_AS i32x4v*)(l),     0, 0);
  __builtin_amdgcn_global_load_async_to_lds_b128(
      (GLOBAL_AS i32x4v*)(g + 8), (LDS_AS i32x4v*)(l + 8), 0, 0);
#else
  const uint4* gp = (const uint4*)g;
  uint4 p0 = gp[0];
  uint4 p1 = gp[1];
  unsigned int* lw = (unsigned int*)l;
  lw[0]=p0.x; lw[1]=p0.y; lw[2]=p0.z; lw[3]=p0.w;
  lw[4]=p1.x; lw[5]=p1.y; lw[6]=p1.z; lw[7]=p1.w;
#endif
}

__device__ __forceinline__ void wait_stage_pending2() {
#if HAVE_ASYNC_LDS
#if __has_builtin(__builtin_amdgcn_s_wait_asynccnt)
  __builtin_amdgcn_s_wait_asynccnt(2);
#else
  asm volatile("s_wait_asynccnt 2" ::: "memory");
#endif
#endif
}
__device__ __forceinline__ void wait_stage_pending0() {
#if HAVE_ASYNC_LDS
#if __has_builtin(__builtin_amdgcn_s_wait_asynccnt)
  __builtin_amdgcn_s_wait_asynccnt(0);
#else
  asm volatile("s_wait_asynccnt 0" ::: "memory");
#endif
#endif
}

// ---------------------------------------------------------------------------
// Pack an f32 matrix into the CDNA5 WMMA B-operand (32x16 bf16) fragment
// layout.  B[k][n]: element i of lane half hs holds K = 8*hs+i (i<8) or
// 16+8*hs+(i-8) (i>=8), N = lane&15.  Fragment (kt,nt) stored as 32 lanes x 16
// halves contiguous.  TR=true: source is codebook [N x Kd] (B = cb^T).
// ---------------------------------------------------------------------------
template<bool TR>
__global__ __launch_bounds__(256)
void pack_b_kernel(const float* __restrict__ W, unsigned short* __restrict__ out,
                   int Kd, int N)
{
  long t = (long)blockIdx.x * 256 + threadIdx.x;   // one packed element
  long total = (long)Kd * N;
  if (t >= total) return;
  int  i    = (int)(t & 15);
  int  lane = (int)((t >> 4) & 31);
  long frag = t >> 9;
  int  ntiles = N >> 4;
  int  kt = (int)(frag / ntiles);
  int  nt = (int)(frag % ntiles);
  int  vg = i >> 1, h = i & 1, hs = lane >> 4;
  int  k = kt * 32 + (i < 8 ? i : 8 + i) + 8 * hs;
  int  n = nt * 16 + (lane & 15);
  float v = TR ? W[(long)n * Kd + k] : W[(long)k * N + n];
  out[t] = f2bf(v);
}

__global__ __launch_bounds__(256)
void tobf16_kernel(const float* __restrict__ x, unsigned short* __restrict__ o, long n)
{
  long t = (long)blockIdx.x * 256 + threadIdx.x;
  if (t < n) o[t] = f2bf(x[t]);
}

__global__ __launch_bounds__(256)
void cnorm_kernel(const float* __restrict__ cb, float* __restrict__ cn)
{
  int wave = threadIdx.x >> 5, lane = threadIdx.x & 31;
  int row = blockIdx.x * 8 + wave;
  float s = 0.f;
  for (int j = lane; j < LATC; j += 32) { float v = cb[(long)row * LATC + j]; s += v * v; }
  #pragma unroll
  for (int m = 16; m >= 1; m >>= 1) s += __shfl_xor(s, m, 32);
  if (lane == 0) cn[row] = s;
}

// ---------------------------------------------------------------------------
// Generic bf16 WMMA GEMM, 128x64 macro-tile, 8 waves (each: 16 rows x 64 cols)
// Double-buffered LDS A-tile with async global->LDS staging.
// MODE 0: out = bf16(gelu(A*B + bias))
// MODE 1: out = bf16 and f32 of (A*B + bias)          (z_e)
// MODE 2: partial[block] = sum((A*B + bias - Xref)^2) (recon loss, no store)
// ---------------------------------------------------------------------------
#define LSTR 40   // halves; 80B rows: 16B-aligned for async B128, and the 16
                  // rows a half-wave touches map to 16 distinct LDS banks.

template<int MODE>
__global__ __launch_bounds__(256)
void gemm_bf16_wmma(const unsigned short* __restrict__ A,
                    const unsigned short* __restrict__ Bp,
                    const float* __restrict__ bias,
                    unsigned short* __restrict__ outB,
                    float* __restrict__ outF,
                    const float* __restrict__ Xref,
                    float* __restrict__ partial,
                    int M, int N, int K)
{
  __shared__ unsigned short ldsA[2][128 * LSTR];
  __shared__ float swv[8];
  const int tid  = threadIdx.x;
  const int wave = tid >> 5;
  const int lane = tid & 31;
  const int hs   = lane >> 4;
  const int l15  = lane & 15;
  const int rowBlock = blockIdx.y * 128;
  const int colBlock = blockIdx.x * 64;
  const int ntiles = N >> 4;

  f32x8 acc[4] = {};

  const int srow = tid >> 1;          // A staging: 128 rows x 32 halves
  const int scol = (tid & 1) * 16;
  const unsigned short* gbase = A + (size_t)(rowBlock + srow) * K + scol;
  const int lofs = srow * LSTR + scol;

  stage32(gbase, &ldsA[0][lofs]);     // prime tile 0
  int buf = 0;

  for (int k0 = 0; k0 < K; k0 += 32) {
    const bool hasNext = (k0 + 32 < K);
    if (hasNext) {
      stage32(gbase + k0 + 32, &ldsA[buf ^ 1][lofs]);
      wait_stage_pending2();          // tile k0 landed; next tile in flight
    } else {
      wait_stage_pending0();
    }
    __syncthreads();

    // A fragment (16x32 bf16): lane half selects K {0-7,16-23}/{8-15,24-31}
    FragAB af;
    const unsigned short* arow = &ldsA[buf][(wave * 16 + l15) * LSTR];
    #pragma unroll
    for (int vg = 0; vg < 8; ++vg) {
      int k = (vg < 4 ? 2 * vg : 16 + 2 * (vg - 4)) + 8 * hs;
      af.u[vg] = *(const unsigned int*)&arow[k];
    }
    const int kt = k0 >> 5;
    #pragma unroll
    for (int nt = 0; nt < 4; ++nt) {
      FragAB bf_;
      const uint4* bp = (const uint4*)Bp
          + ((size_t)kt * ntiles + (colBlock >> 4) + nt) * 64 + lane * 2;
      uint4 b0 = bp[0];
      uint4 b1 = bp[1];
      bf_.u[0]=b0.x; bf_.u[1]=b0.y; bf_.u[2]=b0.z; bf_.u[3]=b0.w;
      bf_.u[4]=b1.x; bf_.u[5]=b1.y; bf_.u[6]=b1.z; bf_.u[7]=b1.w;
      acc[nt] = __builtin_amdgcn_wmma_f32_16x16x32_bf16(
          false, af.b, false, bf_.b, (short)0, acc[nt], false, false);
    }
    __syncthreads();                  // all reads done before buffer reuse
    buf ^= 1;
  }

  float lsum = 0.0f;
  #pragma unroll
  for (int nt = 0; nt < 4; ++nt) {
    const int col = colBlock + nt * 16 + l15;
    const float bv = bias[col];
    #pragma unroll
    for (int r = 0; r < 8; ++r) {
      const int row = rowBlock + wave * 16 + r + 8 * hs;  // C: M=r+8*hs, N=l15
      float v = acc[nt][r] + bv;
      if (MODE == 0) {
        v = gelu_exact(v);
        outB[(size_t)row * N + col] = f2bf(v);
      } else if (MODE == 1) {
        outB[(size_t)row * N + col] = f2bf(v);
        outF[(size_t)row * N + col] = v;
      } else {
        float d = v - Xref[(size_t)row * N + col];
        lsum += d * d;
      }
    }
  }
  if (MODE == 2) {
    #pragma unroll
    for (int m = 16; m >= 1; m >>= 1) lsum += __shfl_xor(lsum, m, 32);
    if (lane == 0) swv[wave] = lsum;
    __syncthreads();
    if (tid == 0) {
      float s = 0.f;
      #pragma unroll
      for (int w = 0; w < 8; ++w) s += swv[w];
      partial[blockIdx.y * gridDim.x + blockIdx.x] = s;
    }
  }
}

// ---------------------------------------------------------------------------
// Fused distance + argmin: 128 rows of z_e resident in LDS, sweep 4096 codes.
// score = ||c||^2 - 2*z.c  (||z||^2 is row-constant, irrelevant for argmin)
// B fragments are software-pipelined (ping-pong register sets) so the
// (kt+1) loads are in flight while the four WMMAs of kt execute.
// ---------------------------------------------------------------------------
#define ZSTR 258

__device__ __forceinline__ void load_bfrags(FragAB* dst, const unsigned short* CBp,
                                            int kt, int nb, int lane)
{
  #pragma unroll
  for (int nt = 0; nt < 4; ++nt) {
    const uint4* bp = (const uint4*)CBp
        + ((size_t)kt * (KCODES / 16) + nb * 4 + nt) * 64 + lane * 2;
    uint4 b0 = bp[0], b1 = bp[1];
    dst[nt].u[0]=b0.x; dst[nt].u[1]=b0.y; dst[nt].u[2]=b0.z; dst[nt].u[3]=b0.w;
    dst[nt].u[4]=b1.x; dst[nt].u[5]=b1.y; dst[nt].u[6]=b1.z; dst[nt].u[7]=b1.w;
  }
}

__global__ __launch_bounds__(256)
void vq_argmin(const unsigned short* __restrict__ ZEb,
               const unsigned short* __restrict__ CBp,
               const float* __restrict__ cnorm,
               int* __restrict__ idxOut)
{
  extern __shared__ unsigned short ldsZ[];   // 128 x 258 halves
  const int tid  = threadIdx.x;
  const int wave = tid >> 5;
  const int lane = tid & 31;
  const int hs   = lane >> 4;
  const int l15  = lane & 15;
  const int rowBlock = blockIdx.x * 128;

  #pragma unroll
  for (int j = 0; j < 8; ++j) {              // stage 128x256 bf16
    int c = j * 256 + tid;
    int row = c >> 4;
    int colOff = (c & 15) * 16;
    const uint4* g = (const uint4*)(ZEb + (size_t)(rowBlock + row) * LATC + colOff);
    uint4 p0 = g[0], p1 = g[1];
    unsigned int* lw = (unsigned int*)&ldsZ[row * ZSTR + colOff];
    lw[0]=p0.x; lw[1]=p0.y; lw[2]=p0.z; lw[3]=p0.w;
    lw[4]=p1.x; lw[5]=p1.y; lw[6]=p1.z; lw[7]=p1.w;
  }
  __syncthreads();

  float bestv[8];
  int   besti[8];
  #pragma unroll
  for (int r = 0; r < 8; ++r) { bestv[r] = 3.4e38f; besti[r] = 0; }

  FragAB afr[8];                              // preload all K fragments (K=256)
  #pragma unroll
  for (int kt = 0; kt < 8; ++kt) {
    const unsigned short* arow = &ldsZ[(wave * 16 + l15) * ZSTR + kt * 32];
    #pragma unroll
    for (int vg = 0; vg < 8; ++vg) {
      int k = (vg < 4 ? 2 * vg : 16 + 2 * (vg - 4)) + 8 * hs;
      afr[kt].u[vg] = *(const unsigned int*)&arow[k];
    }
  }

  for (int nb = 0; nb < KCODES / 64; ++nb) {
    f32x8 acc[4] = {};
    FragAB bb[2][4];
    load_bfrags(bb[0], CBp, 0, nb, lane);     // prime kt = 0
    #pragma unroll
    for (int kt = 0; kt < 8; ++kt) {
      if (kt < 7)
        load_bfrags(bb[(kt + 1) & 1], CBp, kt + 1, nb, lane);  // next in flight
      #pragma unroll
      for (int nt = 0; nt < 4; ++nt) {
        acc[nt] = __builtin_amdgcn_wmma_f32_16x16x32_bf16(
            false, afr[kt].b, false, bb[kt & 1][nt].b, (short)0, acc[nt],
            false, false);
      }
    }
    #pragma unroll
    for (int nt = 0; nt < 4; ++nt) {
      int n = nb * 64 + nt * 16 + l15;
      float cn = cnorm[n];
      #pragma unroll
      for (int r = 0; r < 8; ++r) {
        float v = cn - 2.0f * acc[nt][r];
        if (v < bestv[r]) { bestv[r] = v; besti[r] = n; }   // strict <: first idx wins
      }
    }
  }

  #pragma unroll
  for (int m = 1; m <= 8; m <<= 1) {          // min-reduce within 16-lane half
    #pragma unroll
    for (int r = 0; r < 8; ++r) {
      float ov = __shfl_xor(bestv[r], m, 32);
      int   oi = __shfl_xor(besti[r], m, 32);
      if (ov < bestv[r] || (ov == bestv[r] && oi < besti[r])) { bestv[r] = ov; besti[r] = oi; }
    }
  }
  if (l15 == 0) {
    #pragma unroll
    for (int r = 0; r < 8; ++r)
      idxOut[rowBlock + wave * 16 + r + 8 * hs] = besti[r];
  }
}

// ---------------------------------------------------------------------------
// Gather z_q (f32 -> d_out, bf16 -> decoder) + commitment partial per row
// ---------------------------------------------------------------------------
__global__ __launch_bounds__(256)
void gather_commit(const int* __restrict__ idx, const float* __restrict__ cb,
                   const float* __restrict__ zef, float* __restrict__ zlat,
                   unsigned short* __restrict__ zqb, float* __restrict__ pcommit)
{
  __shared__ float sw[8];
  int row = blockIdx.x;
  int t = threadIdx.x;
  int code = idx[row];
  float v = cb[(size_t)code * LATC + t];
  zlat[(size_t)row * LATC + t] = v;
  zqb [(size_t)row * LATC + t] = f2bf(v);
  float d = v - zef[(size_t)row * LATC + t];
  float s = d * d;
  #pragma unroll
  for (int m = 16; m >= 1; m >>= 1) s += __shfl_xor(s, m, 32);
  int lane = t & 31, wave = t >> 5;
  if (lane == 0) sw[wave] = s;
  __syncthreads();
  if (t == 0) {
    float tot = 0.f;
    #pragma unroll
    for (int w = 0; w < 8; ++w) tot += sw[w];
    pcommit[row] = tot;
  }
}

__global__ __launch_bounds__(256)
void finalize_loss(const float* __restrict__ prec, int nrec,
                   const float* __restrict__ pcom, int ncom,
                   float* __restrict__ lossOut)
{
  __shared__ float sh[256];
  int t = threadIdx.x;
  float s1 = 0.f, s2 = 0.f;
  for (int i = t; i < nrec; i += 256) s1 += prec[i];
  for (int i = t; i < ncom; i += 256) s2 += pcom[i];
  sh[t] = s1; __syncthreads();
  for (int step = 128; step >= 1; step >>= 1) { if (t < step) sh[t] += sh[t + step]; __syncthreads(); }
  float recon = sh[0];
  __syncthreads();
  sh[t] = s2; __syncthreads();
  for (int step = 128; step >= 1; step >>= 1) { if (t < step) sh[t] += sh[t + step]; __syncthreads(); }
  if (t == 0) {
    float commit = sh[0];
    lossOut[0] = recon / ((float)B_SZ * (float)F_DIMC)
               + 0.5f * (commit / ((float)B_SZ * (float)LATC));
  }
}

// ---------------------------------------------------------------------------
extern "C" void kernel_launch(void* const* d_in, const int* in_sizes, int n_in,
                              void* d_out, int out_size, void* d_ws, size_t ws_size,
                              hipStream_t stream)
{
  const float* x      = (const float*)d_in[0];
  const float* enc_w1 = (const float*)d_in[1];
  const float* enc_b1 = (const float*)d_in[2];
  const float* enc_w2 = (const float*)d_in[3];
  const float* enc_b2 = (const float*)d_in[4];
  const float* lat_w  = (const float*)d_in[5];
  const float* lat_b  = (const float*)d_in[6];
  const float* cb     = (const float*)d_in[7];
  const float* dec_w1 = (const float*)d_in[8];
  const float* dec_b1 = (const float*)d_in[9];
  const float* dec_w2 = (const float*)d_in[10];
  const float* dec_b2 = (const float*)d_in[11];
  const float* out_w  = (const float*)d_in[12];
  const float* out_b  = (const float*)d_in[13];

  char* ws = (char*)d_ws;
  size_t off = 0;
  auto alloc = [&](size_t bytes) -> void* {
    void* p = ws + off;
    off = (off + bytes + 255) & ~(size_t)255;
    return p;
  };
  unsigned short* XB   = (unsigned short*)alloc((size_t)B_SZ * F_DIMC * 2);
  unsigned short* W1P  = (unsigned short*)alloc((size_t)F_DIMC * 64 * 2);
  unsigned short* W2P  = (unsigned short*)alloc((size_t)64 * HIDC * 2);
  unsigned short* W3P  = (unsigned short*)alloc((size_t)HIDC * LATC * 2);
  unsigned short* CBP  = (unsigned short*)alloc((size_t)LATC * KCODES * 2);
  unsigned short* DW1P = (unsigned short*)alloc((size_t)LATC * 64 * 2);
  unsigned short* DW2P = (unsigned short*)alloc((size_t)64 * HIDC * 2);
  unsigned short* OWP  = (unsigned short*)alloc((size_t)HIDC * F_DIMC * 2);
  float*          CN   = (float*)alloc((size_t)KCODES * 4);
  unsigned short* H1   = (unsigned short*)alloc((size_t)B_SZ * 64 * 2);
  unsigned short* H2   = (unsigned short*)alloc((size_t)B_SZ * HIDC * 2);
  unsigned short* ZEB  = (unsigned short*)alloc((size_t)B_SZ * LATC * 2);
  float*          ZEF  = (float*)alloc((size_t)B_SZ * LATC * 4);
  int*            IDX  = (int*)alloc((size_t)B_SZ * 4);
  unsigned short* ZQB  = (unsigned short*)alloc((size_t)B_SZ * LATC * 2);
  unsigned short* D1   = (unsigned short*)alloc((size_t)B_SZ * 64 * 2);
  unsigned short* D2   = (unsigned short*)alloc((size_t)B_SZ * HIDC * 2);
  float*          PREC = (float*)alloc((size_t)2048 * 4);
  float*          PCOM = (float*)alloc((size_t)B_SZ * 4);

  float* zlat    = (float*)d_out;
  float* lossOut = (float*)d_out + (size_t)B_SZ * LATC;

  // --- pack / convert ---
  tobf16_kernel<<<(B_SZ * F_DIMC) / 256, 256, 0, stream>>>(x, XB, (long)B_SZ * F_DIMC);
  pack_b_kernel<false><<<(F_DIMC * 64) / 256, 256, 0, stream>>>(enc_w1, W1P, F_DIMC, 64);
  pack_b_kernel<false><<<(64 * HIDC) / 256, 256, 0, stream>>>(enc_w2, W2P, 64, HIDC);
  pack_b_kernel<false><<<(HIDC * LATC) / 256, 256, 0, stream>>>(lat_w, W3P, HIDC, LATC);
  pack_b_kernel<true ><<<(LATC * KCODES) / 256, 256, 0, stream>>>(cb, CBP, LATC, KCODES);
  pack_b_kernel<false><<<(LATC * 64) / 256, 256, 0, stream>>>(dec_w1, DW1P, LATC, 64);
  pack_b_kernel<false><<<(64 * HIDC) / 256, 256, 0, stream>>>(dec_w2, DW2P, 64, HIDC);
  pack_b_kernel<false><<<(HIDC * F_DIMC) / 256, 256, 0, stream>>>(out_w, OWP, HIDC, F_DIMC);
  cnorm_kernel<<<KCODES / 8, 256, 0, stream>>>(cb, CN);

  // --- encoder ---
  gemm_bf16_wmma<0><<<dim3(1, B_SZ / 128), 256, 0, stream>>>(
      XB, W1P, enc_b1, H1, nullptr, nullptr, nullptr, B_SZ, 64, F_DIMC);
  gemm_bf16_wmma<0><<<dim3(HIDC / 64, B_SZ / 128), 256, 0, stream>>>(
      H1, W2P, enc_b2, H2, nullptr, nullptr, nullptr, B_SZ, HIDC, 64);
  gemm_bf16_wmma<1><<<dim3(LATC / 64, B_SZ / 128), 256, 0, stream>>>(
      H2, W3P, lat_b, ZEB, ZEF, nullptr, nullptr, B_SZ, LATC, HIDC);

  // --- quantize ---
  vq_argmin<<<B_SZ / 128, 256, 128 * ZSTR * 2, stream>>>(ZEB, CBP, CN, IDX);
  gather_commit<<<B_SZ, 256, 0, stream>>>(IDX, cb, ZEF, zlat, ZQB, PCOM);

  // --- decoder ---
  gemm_bf16_wmma<0><<<dim3(1, B_SZ / 128), 256, 0, stream>>>(
      ZQB, DW1P, dec_b1, D1, nullptr, nullptr, nullptr, B_SZ, 64, LATC);
  gemm_bf16_wmma<0><<<dim3(HIDC / 64, B_SZ / 128), 256, 0, stream>>>(
      D1, DW2P, dec_b2, D2, nullptr, nullptr, nullptr, B_SZ, HIDC, 64);
  gemm_bf16_wmma<2><<<dim3(F_DIMC / 64, B_SZ / 128), 256, 0, stream>>>(
      D2, OWP, out_b, nullptr, nullptr, x, PREC, B_SZ, F_DIMC, HIDC);

  // --- loss ---
  finalize_loss<<<1, 256, 0, stream>>>(PREC, (F_DIMC / 64) * (B_SZ / 128),
                                       PCOM, B_SZ, lossOut);
}
